// ScaledDotProductAttention_77378130805387
// MI455X (gfx1250) — matrix operand order repaired and checked by reading, not compiled, not run on hardware
//
#include <hip/hip_runtime.h>

typedef __bf16 bf16;
typedef __attribute__((ext_vector_type(16))) __bf16 v16bf;
typedef __attribute__((ext_vector_type(8)))  __bf16 v8bf;
typedef __attribute__((ext_vector_type(4)))  __bf16 v4bf;
typedef __attribute__((ext_vector_type(8)))  float  v8f;

static constexpr int NB = 16;    // batch
static constexpr int SS = 2048;  // sequence
static constexpr int HH = 1024;  // hidden

#define CAT16(a, b) __builtin_shufflevector(a, b, 0,1,2,3,4,5,6,7,8,9,10,11,12,13,14,15)
#define WMMA_BF16(A, B, C) \
  __builtin_amdgcn_wmma_f32_16x16x32_bf16(false, A, false, B, (short)0, C, false, false)

// CDNA5 async global->LDS copy (ASYNCcnt-tracked, no VGPR data staging).
__device__ inline void async_copy_b128(unsigned lds_off, const void* gptr) {
  asm volatile("global_load_async_to_lds_b128 %0, %1, off"
               :: "v"(lds_off), "v"((unsigned long long)(size_t)gptr)
               : "memory");
}
__device__ inline void wait_asynccnt0() {
  asm volatile("s_wait_asynccnt 0" ::: "memory");
}

// Load a 16-element bf16 operand as two b128 chunks at byte offsets off, off+d2.
__device__ inline v16bf ldpair_g(const char* __restrict__ base, unsigned off, unsigned d2) {
  v8bf x0 = *(const v8bf*)(base + off);
  v8bf x1 = *(const v8bf*)(base + off + d2);
  return CAT16(x0, x1);
}
__device__ inline v16bf ldpair_lds(const bf16* p) {  // LDS: chunks at +0 and +8 elems
  v8bf x0 = *(const v8bf*)(p);
  v8bf x1 = *(const v8bf*)(p + 8);
  return CAT16(x0, x1);
}

// ---------------------------------------------------------------- fp32 -> bf16
__global__ void k_cvt_bf16(const float* __restrict__ src, bf16* __restrict__ dst, int n) {
  int i = (blockIdx.x * blockDim.x + threadIdx.x) * 4;
  if (i + 3 < n) {
    float4 f = *(const float4*)(src + i);
    v4bf o;
    o[0] = (bf16)f.x; o[1] = (bf16)f.y; o[2] = (bf16)f.z; o[3] = (bf16)f.w;
    *(v4bf*)(dst + i) = o;
  }
}

__global__ void k_zero_f32(float* __restrict__ p, int n) {
  int i = blockIdx.x * blockDim.x + threadIdx.x;
  if (i < n) p[i] = 0.0f;
}

// ------------------------------------------------- QKV projection: Y = x @ W^T
// Block = 256 threads (8 waves). Block tile = 256 rows x 64 cols.
// Wave: 32 rows x 64 cols = 8 WMMA tiles per k-step; 12 b128 loads / 8 WMMAs.
// Software-pipelined: next k-step's operands load while current WMMAs execute.
__global__ __launch_bounds__(256)
void k_gemm_qkv(const bf16* __restrict__ X,
                const bf16* __restrict__ Wq, const bf16* __restrict__ Wk,
                const bf16* __restrict__ Wv,
                bf16* __restrict__ Qo, bf16* __restrict__ Ko, bf16* __restrict__ Vo) {
  const bf16* W = (blockIdx.z == 0) ? Wq : ((blockIdx.z == 1) ? Wk : Wv);
  bf16*       Y = (blockIdx.z == 0) ? Qo : ((blockIdx.z == 1) ? Ko : Vo);

  const int wave = threadIdx.x >> 5;
  const int lane = threadIdx.x & 31;
  const int half = lane >> 4;        // 0: lanes 0-15, 1: lanes 16-31
  const int ln   = lane & 15;

  const int rowBase = blockIdx.y * 256 + wave * 32;
  const int n0      = blockIdx.x * 64;

  const char* Xc = (const char*)X;
  const char* Wc = (const char*)W;
  const unsigned rstride = HH * 2u;  // row stride in bytes (2 KB)

  // 32-bit byte offsets (buffers < 128 MB): A chunks at +0/+16B, B at +0/+16B
  const unsigned aoff0 = (unsigned)(rowBase + ln) * rstride + (half ? 32u : 0u);
  const unsigned aoff1 = aoff0 + 16u * rstride;
  unsigned boff[4];
#pragma unroll
  for (int j = 0; j < 4; ++j)
    boff[j] = (unsigned)(n0 + 16 * j + ln) * rstride + (half ? 32u : 0u);

  v8f acc[8] = {v8f{}, v8f{}, v8f{}, v8f{}, v8f{}, v8f{}, v8f{}, v8f{}};

  // prologue: stage k-step 0
  v16bf A0 = ldpair_g(Xc, aoff0, 16);
  v16bf A1 = ldpair_g(Xc, aoff1, 16);
  v16bf B0 = ldpair_g(Wc, boff[0], 16);
  v16bf B1 = ldpair_g(Wc, boff[1], 16);
  v16bf B2 = ldpair_g(Wc, boff[2], 16);
  v16bf B3 = ldpair_g(Wc, boff[3], 16);

  for (int dk = 32; dk <= HH; dk += 32) {
    const unsigned d2 = (dk < HH) ? (unsigned)dk * 2u : 0u;  // last iter: dummy reload
    v16bf A0n = ldpair_g(Xc, aoff0 + d2, 16);
    v16bf A1n = ldpair_g(Xc, aoff1 + d2, 16);
    v16bf B0n = ldpair_g(Wc, boff[0] + d2, 16);
    v16bf B1n = ldpair_g(Wc, boff[1] + d2, 16);
    v16bf B2n = ldpair_g(Wc, boff[2] + d2, 16);
    v16bf B3n = ldpair_g(Wc, boff[3] + d2, 16);
    acc[0] = WMMA_BF16(A0, B0, acc[0]);
    acc[4] = WMMA_BF16(A1, B0, acc[4]);
    acc[1] = WMMA_BF16(A0, B1, acc[1]);
    acc[5] = WMMA_BF16(A1, B1, acc[5]);
    acc[2] = WMMA_BF16(A0, B2, acc[2]);
    acc[6] = WMMA_BF16(A1, B2, acc[6]);
    acc[3] = WMMA_BF16(A0, B3, acc[3]);
    acc[7] = WMMA_BF16(A1, B3, acc[7]);
    A0 = A0n; A1 = A1n; B0 = B0n; B1 = B1n; B2 = B2n; B3 = B3n;
  }

  // C layout: VGPR v -> row (v + 8*half), col = ln
#pragma unroll
  for (int j = 0; j < 4; ++j) {
#pragma unroll
    for (int v = 0; v < 8; ++v) {
      Y[(size_t)(rowBase + v + 8 * half) * HH + (n0 + 16 * j + ln)] = (bf16)acc[j][v];
      Y[(size_t)(rowBase + 16 + v + 8 * half) * HH + (n0 + 16 * j + ln)] = (bf16)acc[j + 4][v];
    }
  }
}

// ----------------------------------------------- attention pass 1: row max/sum
// grid (128 q-tiles, B). Block 256 (8 waves); wave owns a 32-key strip per
// iteration (2 WMMA tiles sharing one LDS A operand), 8 iterations cover S.
__global__ __launch_bounds__(256)
void k_attn_stats(const bf16* __restrict__ Q, const bf16* __restrict__ K,
                  float* __restrict__ Mg, float* __restrict__ Lg) {
  __shared__ bf16  Qs[16 * HH];          // 32 KB: the 16 q-rows for this block
  __shared__ float mred[8][16];
  __shared__ float lred[8][16];

  const int qt = blockIdx.x, b = blockIdx.y;
  const int tid = threadIdx.x;

  {  // async global->LDS copy of the contiguous 32 KB Q tile
    const bf16* gsrc = Q + ((size_t)b * SS + qt * 16) * HH;
    const unsigned lds0 = (unsigned)(size_t)(&Qs[0]);
#pragma unroll
    for (int c = 0; c < 8; ++c) {
      const int chunk = tid + 256 * c;           // 2048 x 16B
      async_copy_b128(lds0 + chunk * 16, (const char*)gsrc + chunk * 16);
    }
    wait_asynccnt0();
  }
  __syncthreads();

  const int wave = tid >> 5, lane = tid & 31;
  const int half = lane >> 4, ln = lane & 15;
  const float sc = 0.03125f;  // 1/sqrt(1024)
  const char* Kc = (const char*)K;
  const bf16* qrow = Qs + ln * HH + (half ? 16 : 0);

  float mrun[8], lrun[8];
#pragma unroll
  for (int v = 0; v < 8; ++v) { mrun[v] = -1e30f; lrun[v] = 0.0f; }

  for (int it = 0; it < 8; ++it) {
    const int k0 = (it * 8 + wave) * 32;
    const unsigned koff0 =
        (unsigned)((b * SS + k0 + ln) * HH + (half ? 16 : 0)) * 2u;
    const unsigned koff1 = koff0 + 16u * HH * 2u;
    v8f e0{}, e1{};
    v16bf B0 = ldpair_g(Kc, koff0, 16);
    v16bf B1 = ldpair_g(Kc, koff1, 16);
    for (int dk = 32; dk <= HH; dk += 32) {
      const unsigned d2 = (dk < HH) ? (unsigned)dk * 2u : 0u;
      __builtin_prefetch(Kc + koff0 + d2 + 2048, 0, 1);
      v16bf B0n = ldpair_g(Kc, koff0 + d2, 16);
      v16bf B1n = ldpair_g(Kc, koff1 + d2, 16);
      v16bf A = ldpair_lds(qrow + dk - 32);
      e0 = WMMA_BF16(A, B0, e0);
      e1 = WMMA_BF16(A, B1, e1);
      B0 = B0n; B1 = B1n;
    }
    // online max/sumexp per row over the 32-key strip
#pragma unroll
    for (int v = 0; v < 8; ++v) {
      float ev0 = e0[v] * sc, ev1 = e1[v] * sc;
      float tm = fmaxf(ev0, ev1);
      tm = fmaxf(tm, __shfl_xor(tm, 1, 32));
      tm = fmaxf(tm, __shfl_xor(tm, 2, 32));
      tm = fmaxf(tm, __shfl_xor(tm, 4, 32));
      tm = fmaxf(tm, __shfl_xor(tm, 8, 32));
      float mn = fmaxf(mrun[v], tm);
      float p = __expf(ev0 - mn) + __expf(ev1 - mn);
      p += __shfl_xor(p, 1, 32);
      p += __shfl_xor(p, 2, 32);
      p += __shfl_xor(p, 4, 32);
      p += __shfl_xor(p, 8, 32);
      lrun[v] = lrun[v] * __expf(mrun[v] - mn) + p;
      mrun[v] = mn;
    }
  }

  if (ln == 0) {
#pragma unroll
    for (int v = 0; v < 8; ++v) {
      mred[wave][v + 8 * half] = mrun[v];
      lred[wave][v + 8 * half] = lrun[v];
    }
  }
  __syncthreads();

  if (tid < 16) {
    float mf = -1e30f;
#pragma unroll
    for (int w = 0; w < 8; ++w) mf = fmaxf(mf, mred[w][tid]);
    float lf = 0.0f;
#pragma unroll
    for (int w = 0; w < 8; ++w) lf += lred[w][tid] * __expf(mred[w][tid] - mf);
    Mg[b * SS + qt * 16 + tid] = mf;
    Lg[b * SS + qt * 16 + tid] = lf;
  }
}

// --------------------------------- attention pass 2: softmax column sums (B,S)
__global__ __launch_bounds__(256)
void k_attn_colsum(const bf16* __restrict__ Q, const bf16* __restrict__ K,
                   const float* __restrict__ Mg, const float* __restrict__ Lg,
                   float* __restrict__ aw) {
  __shared__ bf16  Qs[16 * HH];
  __shared__ float Ms[16];
  __shared__ float Li[16];

  const int qt = blockIdx.x, b = blockIdx.y;
  const int tid = threadIdx.x;

  {
    const bf16* gsrc = Q + ((size_t)b * SS + qt * 16) * HH;
    const unsigned lds0 = (unsigned)(size_t)(&Qs[0]);
#pragma unroll
    for (int c = 0; c < 8; ++c) {
      const int chunk = tid + 256 * c;
      async_copy_b128(lds0 + chunk * 16, (const char*)gsrc + chunk * 16);
    }
    wait_asynccnt0();
  }
  if (tid < 16) {
    Ms[tid] = Mg[b * SS + qt * 16 + tid];
    Li[tid] = 1.0f / Lg[b * SS + qt * 16 + tid];
  }
  __syncthreads();

  const int wave = tid >> 5, lane = tid & 31;
  const int half = lane >> 4, ln = lane & 15;
  const float sc = 0.03125f;
  const float invS = 1.0f / (float)SS;
  const char* Kc = (const char*)K;
  const bf16* qrow = Qs + ln * HH + (half ? 16 : 0);

  for (int it = 0; it < 8; ++it) {
    const int k0 = (it * 8 + wave) * 32;
    const unsigned koff0 =
        (unsigned)((b * SS + k0 + ln) * HH + (half ? 16 : 0)) * 2u;
    const unsigned koff1 = koff0 + 16u * HH * 2u;
    v8f e0{}, e1{};
    v16bf B0 = ldpair_g(Kc, koff0, 16);
    v16bf B1 = ldpair_g(Kc, koff1, 16);
    for (int dk = 32; dk <= HH; dk += 32) {
      const unsigned d2 = (dk < HH) ? (unsigned)dk * 2u : 0u;
      __builtin_prefetch(Kc + koff0 + d2 + 2048, 0, 1);
      v16bf B0n = ldpair_g(Kc, koff0 + d2, 16);
      v16bf B1n = ldpair_g(Kc, koff1 + d2, 16);
      v16bf A = ldpair_lds(qrow + dk - 32);
      e0 = WMMA_BF16(A, B0, e0);
      e1 = WMMA_BF16(A, B1, e1);
      B0 = B0n; B1 = B1n;
    }
    float s0 = 0.0f, s1 = 0.0f;
#pragma unroll
    for (int v = 0; v < 8; ++v) {
      const int row = v + 8 * half;
      const float w = Li[row] * invS;
      s0 += __expf(e0[v] * sc - Ms[row]) * w;
      s1 += __expf(e1[v] * sc - Ms[row]) * w;
    }
    s0 += __shfl_xor(s0, 16, 32);           // combine the two row-halves
    s1 += __shfl_xor(s1, 16, 32);
    if (half == 0) {
      atomicAdd(&aw[b * SS + k0 + ln], s0);
      atomicAdd(&aw[b * SS + k0 + 16 + ln], s1);
    }
  }
}

// ------------- context[b,d] = sum_k aw[b,k] * V[b,k,d]  (mean folded into aw)
__global__ __launch_bounds__(128)
void k_context(const float* __restrict__ aw, const bf16* __restrict__ V,
               float* __restrict__ ctx) {
  const int b = blockIdx.y;
  const int d = blockIdx.x * 128 + threadIdx.x;
  const bf16* vb = V + (size_t)b * SS * HH + d;
  const float* a = aw + b * SS;
  float acc = 0.0f;
#pragma unroll 4
  for (int k = 0; k < SS; ++k) acc += a[k] * (float)vb[(size_t)k * HH];
  ctx[b * HH + d] = acc;
}

// ---------------------------------------------------------------------- launch
extern "C" void kernel_launch(void* const* d_in, const int* in_sizes, int n_in,
                              void* d_out, int out_size, void* d_ws, size_t ws_size,
                              hipStream_t stream) {
  const float* x  = (const float*)d_in[0];
  const float* Wq = (const float*)d_in[1];
  const float* Wk = (const float*)d_in[2];
  const float* Wv = (const float*)d_in[3];

  float* ctx = (float*)d_out;              // (B,H)
  float* aw  = (float*)d_out + NB * HH;    // (B,S)

  char* ws = (char*)d_ws;
  size_t off = 0;
  const size_t xbytes = (size_t)NB * SS * HH * sizeof(bf16);
  const size_t wbytes = (size_t)HH * HH * sizeof(bf16);
  bf16* xb  = (bf16*)(ws + off); off += xbytes;
  bf16* wqb = (bf16*)(ws + off); off += wbytes;
  bf16* wkb = (bf16*)(ws + off); off += wbytes;
  bf16* wvb = (bf16*)(ws + off); off += wbytes;
  bf16* Qb  = (bf16*)(ws + off); off += xbytes;
  bf16* Kb  = (bf16*)(ws + off); off += xbytes;
  bf16* Vb  = (bf16*)(ws + off); off += xbytes;
  float* Mg = (float*)(ws + off); off += (size_t)NB * SS * sizeof(float);
  float* Lg = (float*)(ws + off); off += (size_t)NB * SS * sizeof(float);

  // fp32 -> bf16 conversions
  const int nx = NB * SS * HH;
  const int nw = HH * HH;
  k_cvt_bf16<<<nx / 1024, 256, 0, stream>>>(x, xb, nx);
  k_cvt_bf16<<<nw / 1024, 256, 0, stream>>>(Wq, wqb, nw);
  k_cvt_bf16<<<nw / 1024, 256, 0, stream>>>(Wk, wkb, nw);
  k_cvt_bf16<<<nw / 1024, 256, 0, stream>>>(Wv, wvb, nw);
  // attn_weights accumulates via atomics -> zero every launch
  k_zero_f32<<<(NB * SS) / 256, 256, 0, stream>>>(aw, NB * SS);

  // QKV projections (bf16 WMMA, fp32 accumulate)
  dim3 gg(HH / 64, (NB * SS) / 256, 3);
  k_gemm_qkv<<<gg, 256, 0, stream>>>(xb, wqb, wkb, wvb, Qb, Kb, Vb);

  // attention
  dim3 ga(SS / 16, NB);
  k_attn_stats<<<ga, 256, 0, stream>>>(Qb, Kb, Mg, Lg);
  k_attn_colsum<<<ga, 256, 0, stream>>>(Qb, Kb, Mg, Lg, aw);

  // context
  dim3 gc(HH / 128, NB);
  k_context<<<gc, 128, 0, stream>>>(aw, Vb, ctx);
}
